// AttentionHead_87677462380694
// MI455X (gfx1250) — matrix operand order repaired and checked by reading, not compiled, not run on hardware
//
#include <hip/hip_runtime.h>
#include <hip/hip_bf16.h>

typedef __attribute__((ext_vector_type(16))) _Float16 v16h;
typedef __attribute__((ext_vector_type(8)))  _Float16 v8h;
typedef __attribute__((ext_vector_type(4)))  _Float16 v4h;
typedef __attribute__((ext_vector_type(8)))  float    v8f;

union AFrag { v16h v; _Float16 e[16]; };
union CFrag { v8f  v; float     e[8];  };
union HPack { v8h  v; _Float16 e[8];  };

#define EMBED 1024
#define HD    128
#define SEQ   4096

static __device__ __forceinline__ v4h cvt4(float4 f) {
    v4h h;
    h.x = (_Float16)f.x; h.y = (_Float16)f.y;
    h.z = (_Float16)f.z; h.w = (_Float16)f.w;
    return h;
}

// ---------------------------------------------------------------------------
// Kernel 1: QKV projection.  out = x @ W + b  for W in {Wq,Wk,Wv} (grid.y).
// Block = 256 threads (8 waves); each wave 16 rows x 128 cols; block 128 rows.
// Outputs are f16: Q,K as [row][128]; V transposed as [batch][d][SEQ].
// ---------------------------------------------------------------------------
__global__ __launch_bounds__(256) void qkv_kernel(
    const float* __restrict__ x,
    const float* __restrict__ Wq, const float* __restrict__ bq,
    const float* __restrict__ Wk, const float* __restrict__ bk,
    const float* __restrict__ Wv, const float* __restrict__ bv,
    _Float16* __restrict__ Qo, _Float16* __restrict__ Ko,
    _Float16* __restrict__ Vt)
{
    const int which = blockIdx.y;
    const float* W    = (which == 0) ? Wq : (which == 1) ? Wk : Wv;
    const float* bias = (which == 0) ? bq : (which == 1) ? bk : bv;

    __shared__ _Float16 Xs[128][32];    // [row][k]            8 KB
    __shared__ _Float16 Ws[128][32];    // [n][k] (transposed) 8 KB

    const int tid  = threadIdx.x;
    const int wave = tid >> 5;
    const int lane = tid & 31;
    const int half = lane >> 4;
    const int lm   = lane & 15;
    const int rowbase = blockIdx.x * 128;

    CFrag c[8];
#pragma unroll
    for (int nf = 0; nf < 8; ++nf)
#pragma unroll
        for (int r = 0; r < 8; ++r) c[nf].e[r] = 0.0f;

    for (int k0 = 0; k0 < EMBED; k0 += 32) {
        // ---- stage x tile 128x32 : 4096 f32 = 1024 float4, 4 per thread ----
#pragma unroll
        for (int i = 0; i < 4; ++i) {
            int e4  = tid + i * 256;
            int r   = e4 >> 3;
            int cc  = (e4 & 7) * 4;
            float4 f = *(const float4*)&x[(size_t)(rowbase + r) * EMBED + (k0 + cc)];
            *(v4h*)&Xs[r][cc] = cvt4(f);
        }
        // ---- stage W tile 32x128 transposed: vector load along n ----
#pragma unroll
        for (int i = 0; i < 4; ++i) {
            int e4  = tid + i * 256;
            int idx = e4 * 4;
            int ck  = idx >> 7;
            int n   = idx & 127;
            float4 f = *(const float4*)&W[(size_t)(k0 + ck) * HD + n];
            Ws[n + 0][ck] = (_Float16)f.x;
            Ws[n + 1][ck] = (_Float16)f.y;
            Ws[n + 2][ck] = (_Float16)f.z;
            Ws[n + 3][ck] = (_Float16)f.w;
        }
        __syncthreads();

        AFrag a;
#pragma unroll
        for (int v = 0; v < 8; ++v) {
            int kk = (v >> 2) * 16 + half * 8 + (v & 3) * 2;
            a.e[2 * v]     = Xs[wave * 16 + lm][kk];
            a.e[2 * v + 1] = Xs[wave * 16 + lm][kk + 1];
        }
#pragma unroll
        for (int nf = 0; nf < 8; ++nf) {
            AFrag b;
#pragma unroll
            for (int v = 0; v < 8; ++v) {
                int kk = (v >> 2) * 16 + half * 8 + (v & 3) * 2;
                b.e[2 * v]     = Ws[nf * 16 + lm][kk];
                b.e[2 * v + 1] = Ws[nf * 16 + lm][kk + 1];
            }
            c[nf].v = __builtin_amdgcn_wmma_f32_16x16x32_f16(
                false, a.v, false, b.v, (short)0, c[nf].v, false, false);
        }
        __syncthreads();
    }

    if (which == 2) {
        // V: transposed f16 output [batch][d][SEQ]; per lane the 8 C-rows are
        // 8 consecutive seq positions -> one 16B store per column fragment.
        int row0 = rowbase + wave * 16 + 8 * half;
        int b    = row0 >> 12;          // /SEQ
        int s0   = row0 & (SEQ - 1);
#pragma unroll
        for (int nf = 0; nf < 8; ++nf) {
            int col = nf * 16 + lm;
            float bb = bias[col];
            HPack p;
#pragma unroll
            for (int r = 0; r < 8; ++r) p.e[r] = (_Float16)(c[nf].e[r] + bb);
            *(v8h*)&Vt[(size_t)b * HD * SEQ + (size_t)col * SEQ + s0] = p.v;
        }
    } else {
        _Float16* out = (which == 0) ? Qo : Ko;
#pragma unroll
        for (int nf = 0; nf < 8; ++nf) {
            int col = nf * 16 + lm;
            float bb = bias[col];
#pragma unroll
            for (int r = 0; r < 8; ++r) {
                int row = rowbase + wave * 16 + r + 8 * half;
                out[(size_t)row * HD + col] = (_Float16)(c[nf].e[r] + bb);
            }
        }
    }
}

// ---------------------------------------------------------------------------
// Kernel 2: causal flash attention over f16 Q/K/V. Block = 64 query rows
// (4 waves x 16 rows), 64-key tiles; staging is pure b128 copies (no cvt,
// no transpose work: V arrives pre-transposed). Online softmax in exp2 domain.
// ---------------------------------------------------------------------------
__global__ __launch_bounds__(128) void attn_kernel(
    const _Float16* __restrict__ Q, const _Float16* __restrict__ K,
    const _Float16* __restrict__ Vt, float* __restrict__ out)
{
    __shared__ _Float16 Ks[64][128];    // [key][d]          16 KB
    __shared__ _Float16 Vs[128][64];    // [d][key]          16 KB
    __shared__ _Float16 Ps[4][16][64];  // per-wave P scratch 8 KB

    const int tid  = threadIdx.x;
    const int wave = tid >> 5;
    const int lane = tid & 31;
    const int half = lane >> 4;
    const int lm   = lane & 15;
    const int qbase = blockIdx.x * 64;
    const size_t boff = (size_t)blockIdx.y * SEQ;                 // Q/K row base
    const size_t bV   = (size_t)blockIdx.y * HD * SEQ;            // Vt base

    // Q fragments: f16 direct, contiguous 16B chunks -> b128 loads
    AFrag qa[4];
    {
        const _Float16* qp = Q + (boff + qbase + wave * 16 + lm) * HD;
#pragma unroll
        for (int dc = 0; dc < 4; ++dc)
#pragma unroll
            for (int v = 0; v < 8; ++v) {
                int kk = dc * 32 + (v >> 2) * 16 + half * 8 + (v & 3) * 2;
                qa[dc].e[2 * v]     = qp[kk];
                qa[dc].e[2 * v + 1] = qp[kk + 1];
            }
    }

    float m[8], l[8];
    CFrag o[8];
#pragma unroll
    for (int r = 0; r < 8; ++r) { m[r] = -__builtin_inff(); l[r] = 0.0f; }
#pragma unroll
    for (int nf = 0; nf < 8; ++nf)
#pragma unroll
        for (int r = 0; r < 8; ++r) o[nf].e[r] = 0.0f;

    const float scale2 = 0.0883883476483184f * 1.4426950408889634f; // /sqrt(128)*log2(e)

    for (int j0 = 0; j0 <= qbase; j0 += 64) {
        // ---- stage K tile [64][128] : 1024 x 16B chunks, 8 per thread ----
#pragma unroll
        for (int i = 0; i < 8; ++i) {
            int chunk = tid + i * 128;
            int kr  = chunk >> 4;
            int c16 = (chunk & 15) * 8;
            *(v8h*)&Ks[kr][c16] =
                *(const v8h*)&K[(boff + j0 + kr) * HD + c16];
        }
        // ---- stage V tile [128][64] from pre-transposed Vt ----
#pragma unroll
        for (int i = 0; i < 8; ++i) {
            int chunk = tid + i * 128;
            int d  = chunk >> 3;
            int cc = (chunk & 7) * 8;
            *(v8h*)&Vs[d][cc] =
                *(const v8h*)&Vt[bV + (size_t)d * SEQ + j0 + cc];
        }
        if (j0 + 64 <= qbase)  // hint next K tile toward the caches
            __builtin_prefetch(&K[(boff + j0 + 64) * HD + (size_t)(tid & 63) * HD], 0, 0);
        __syncthreads();

        // ---- S = Q K^T over 64 keys (four 16-key column fragments) ----
        CFrag s[4];
#pragma unroll
        for (int cf = 0; cf < 4; ++cf)
#pragma unroll
            for (int r = 0; r < 8; ++r) s[cf].e[r] = 0.0f;
#pragma unroll
        for (int dc = 0; dc < 4; ++dc) {
#pragma unroll
            for (int cf = 0; cf < 4; ++cf) {
                AFrag b;
#pragma unroll
                for (int v = 0; v < 8; ++v) {
                    int kk = dc * 32 + (v >> 2) * 16 + half * 8 + (v & 3) * 2;
                    b.e[2 * v]     = Ks[cf * 16 + lm][kk];
                    b.e[2 * v + 1] = Ks[cf * 16 + lm][kk + 1];
                }
                s[cf].v = __builtin_amdgcn_wmma_f32_16x16x32_f16(
                    false, qa[dc].v, false, b.v, (short)0, s[cf].v, false, false);
            }
        }

        // ---- mask + online softmax (rows live in 16-lane halves) ----
#pragma unroll
        for (int r = 0; r < 8; ++r) {
            int row = r + 8 * half;
            int qg  = qbase + wave * 16 + row;
            float sv[4];
            float t = -__builtin_inff();
#pragma unroll
            for (int cf = 0; cf < 4; ++cf) {
                int c = j0 + cf * 16 + lm;
                sv[cf] = (c <= qg) ? s[cf].e[r] * scale2 : -__builtin_inff();
                t = fmaxf(t, sv[cf]);
            }
            t = fmaxf(t, __shfl_xor(t, 1, 32));
            t = fmaxf(t, __shfl_xor(t, 2, 32));
            t = fmaxf(t, __shfl_xor(t, 4, 32));
            t = fmaxf(t, __shfl_xor(t, 8, 32));
            float mnew = fmaxf(m[r], t);
            float rs = 0.0f;
#pragma unroll
            for (int cf = 0; cf < 4; ++cf) {
                float p = exp2f(sv[cf] - mnew);
                rs += p;
                Ps[wave][row][cf * 16 + lm] = (_Float16)p;
            }
            rs += __shfl_xor(rs, 1, 32);
            rs += __shfl_xor(rs, 2, 32);
            rs += __shfl_xor(rs, 4, 32);
            rs += __shfl_xor(rs, 8, 32);
            float alpha = exp2f(m[r] - mnew);
            l[r] = l[r] * alpha + rs;
            m[r] = mnew;
#pragma unroll
            for (int nf = 0; nf < 8; ++nf) o[nf].e[r] *= alpha;
        }

        // ---- O += P (16x64) x V (64x128): relayout P via LDS ----
        AFrag pa[2];
#pragma unroll
        for (int kc = 0; kc < 2; ++kc)
#pragma unroll
            for (int v = 0; v < 8; ++v) {
                int kk = kc * 32 + (v >> 2) * 16 + half * 8 + (v & 3) * 2;
                pa[kc].e[2 * v]     = Ps[wave][lm][kk];
                pa[kc].e[2 * v + 1] = Ps[wave][lm][kk + 1];
            }
#pragma unroll
        for (int nf = 0; nf < 8; ++nf) {
#pragma unroll
            for (int kc = 0; kc < 2; ++kc) {
                AFrag bv;
#pragma unroll
                for (int v = 0; v < 8; ++v) {
                    int kk = kc * 32 + (v >> 2) * 16 + half * 8 + (v & 3) * 2;
                    bv.e[2 * v]     = Vs[nf * 16 + lm][kk];
                    bv.e[2 * v + 1] = Vs[nf * 16 + lm][kk + 1];
                }
                o[nf].v = __builtin_amdgcn_wmma_f32_16x16x32_f16(
                    false, pa[kc].v, false, bv.v, (short)0, o[nf].v, false, false);
            }
        }
        __syncthreads();
    }

    // ---- normalize and write O (f32) ----
#pragma unroll
    for (int r = 0; r < 8; ++r) {
        float inv = 1.0f / l[r];
        size_t row = boff + qbase + wave * 16 + r + 8 * half;
#pragma unroll
        for (int nf = 0; nf < 8; ++nf)
            out[row * HD + nf * 16 + lm] = o[nf].e[r] * inv;
    }
}

// ---------------------------------------------------------------------------
extern "C" void kernel_launch(void* const* d_in, const int* in_sizes, int n_in,
                              void* d_out, int out_size, void* d_ws, size_t ws_size,
                              hipStream_t stream) {
    const float* x  = (const float*)d_in[0];
    const float* Wq = (const float*)d_in[1];
    const float* bq = (const float*)d_in[2];
    const float* Wk = (const float*)d_in[3];
    const float* bk = (const float*)d_in[4];
    const float* Wv = (const float*)d_in[5];
    const float* bv = (const float*)d_in[6];
    float* outp = (float*)d_out;

    const size_t rows = (size_t)4 * SEQ;          // 16384
    _Float16* Qh = (_Float16*)d_ws;               // 4 MB each
    _Float16* Kh = Qh + rows * HD;
    _Float16* Vt = Kh + rows * HD;                // [batch][d][SEQ]

    qkv_kernel<<<dim3(rows / 128, 3), 256, 0, stream>>>(
        x, Wq, bq, Wk, bk, Wv, bv, Qh, Kh, Vt);

    attn_kernel<<<dim3(SEQ / 64, 4), 128, 0, stream>>>(Qh, Kh, Vt, outp);
}
